// TorchCNN_82102594831014
// MI455X (gfx1250) — matrix operand order repaired and checked by reading, not compiled, not run on hardware
//
#include <hip/hip_runtime.h>
#include <hip/hip_bf16.h>

typedef __attribute__((ext_vector_type(16))) _Float16 v16h;
typedef __attribute__((ext_vector_type(8)))  float    v8f;

#define BATCH 256
#define CIN   3
#define HWD   128
#define OC    16
#define KSZ   27          // CIN*3*3
#define PW    63          // pooled width/height
#define FEAT  (OC*PW*PW)  // 63504
#define NH    256
#define FANIN 512
#define NO    10

// LDS float offset for filter-tap K within the staged patch (c*512 + ky*128 + kx)
__host__ __device__ constexpr int koff(int K) {
    return (K / 9) * 512 + ((K % 9) / 3) * 128 + (K % 3);
}

// ---------------------------------------------------------------------------
// Kernel 1: fused conv3x3 (implicit GEMM via WMMA f16) + bias + ReLU + 2x2 maxpool
// Block = (batch b, pooled row py). 8 waves; wave w covers conv x = 16w..16w+15.
// One v_wmma_f32_16x16x32_f16 per conv row (K=27 padded to 32); fragment fills
// branchless; pooled stores are 3 unconditional + 1 guarded (only px=63 can OOB).
// ---------------------------------------------------------------------------
__global__ __launch_bounds__(256) void conv_pool_wmma(
    const float* __restrict__ in, const float* __restrict__ w,
    const float* __restrict__ bias, float* __restrict__ feats)
{
    __shared__ float s_in[CIN * 4 * HWD + 4];   // 6 KB + pad for tile-7 overhang

    const int blk = blockIdx.x;             // b*PW + py
    const int b   = blk / PW;
    const int py  = blk % PW;
    const int tid = threadIdx.x;

    // cooperative float4 stage of input rows iy = 2py..2py+3, all channels
    const float* inb = in + (size_t)b * CIN * HWD * HWD;
    for (int i = tid; i < (CIN * 4 * HWD) / 4; i += 256) {
        const int c  = i >> 7;
        const int r  = (i >> 5) & 3;
        const int x4 = i & 31;
        ((float4*)s_in)[i] =
            ((const float4*)(inb + (size_t)c * HWD * HWD + (2 * py + r) * HWD))[x4];
    }
    __syncthreads();

    const int wave = tid >> 5;
    const int lane = tid & 31;
    const int m    = lane & 15;             // pixel (A) / oc (B,D)
    const bool hi  = (lane >> 4) != 0;      // half-wave select
    const int ox0  = wave * 16;

    // B fragment (weights 32x16): lane<16 holds K=0..7,16..23, lane>=16 K=8..15,24..31 (pad>=27).
    v16h bfrag;
    #pragma unroll
    for (int h = 0; h < 16; ++h) {
        const bool pad = (16 + h >= KSZ);                 // only h>=11 on hi half
        const int  K   = (hi && pad) ? h : (((h < 8) ? h : h + 8) + (hi ? 8 : 0));
        const float raw = w[m * KSZ + K];
        bfrag[h] = (_Float16)((hi && pad) ? 0.0f : raw);
    }

    v8f acc[2];
    #pragma unroll
    for (int r2 = 0; r2 < 2; ++r2) {        // two conv rows oy = 2py + r2
        const int xbase = r2 * HWD + ox0 + m;
        v16h afrag;
        #pragma unroll
        for (int h = 0; h < 16; ++h) {
            const int  K0  = (h < 8) ? h : h + 8;         // hi=0 K
            const bool pad = (K0 + 8 >= KSZ);             // hi=1 pad (h>=11)
            const int  offL = koff(K0);                   // compile-time consts
            const int  offH = pad ? koff(K0) : koff(K0 + 8);
            const float raw = s_in[(hi ? offH : offL) + xbase];
            afrag[h] = (_Float16)((hi && pad) ? 0.0f : raw);
        }
        v8f z = {};
        acc[r2] = __builtin_amdgcn_wmma_f32_16x16x32_f16(
            false, afrag, false, bfrag, (short)0, z, false, false);
    }

    // bias + ReLU + 2x2 maxpool in accumulator layout:
    // D: VGPR r <-> conv x-offset (lane<16 ? r : r+8), lane%16 = oc.
    const float bv = bias[m];
    float p[4];
    #pragma unroll
    for (int j = 0; j < 4; ++j) {
        float v00 = fmaxf(acc[0][2*j]   + bv, 0.0f);
        float v01 = fmaxf(acc[0][2*j+1] + bv, 0.0f);
        float v10 = fmaxf(acc[1][2*j]   + bv, 0.0f);
        float v11 = fmaxf(acc[1][2*j+1] + bv, 0.0f);
        p[j] = fmaxf(fmaxf(v00, v01), fmaxf(v10, v11));
    }
    // px = wave*8 + hi*4 + j; only (wave==7 && hi && j==3) -> px=63 is OOB.
    float* fp = feats + (size_t)b * FEAT + (size_t)m * (PW * PW)
              + (size_t)py * PW + wave * 8 + (hi ? 4 : 0);
    fp[0] = p[0];
    fp[1] = p[1];
    fp[2] = p[2];
    if (!(wave == 7 && hi)) fp[3] = p[3];
}

// ---------------------------------------------------------------------------
// Kernel 2: sparse gather + hidden layer. One block per batch element; the full
// 254 KB feature row lives in CDNA5's 320 KB LDS (dynamic), gathers are ds_load.
// idx/weights (1 MB total) are re-read per block but resident in the 192 MB L2.
// ---------------------------------------------------------------------------
__global__ __launch_bounds__(256) void gather_hidden(
    const float* __restrict__ feats, const int* __restrict__ idx,
    const float* __restrict__ hw, const float* __restrict__ hb,
    float* __restrict__ hidden)
{
    extern __shared__ float s_feat[];       // FEAT floats = 254016 B dynamic LDS
    const int b   = blockIdx.x;
    const int tid = threadIdx.x;

    const int4*   ih = (const int4*)  (idx + (size_t)tid * FANIN);
    const float4* wh = (const float4*)(hw  + (size_t)tid * FANIN);
    __builtin_prefetch(ih, 0, 0);           // global_prefetch while LDS stage runs
    __builtin_prefetch(wh, 0, 0);

    const float4* fb4 = (const float4*)(feats + (size_t)b * FEAT);
    float4* sf4 = (float4*)s_feat;
    for (int i = tid; i < FEAT / 4; i += 256) sf4[i] = fb4[i];
    __syncthreads();

    float acc = 0.0f;
    #pragma unroll 4
    for (int k = 0; k < FANIN / 4; ++k) {
        const int4   i4 = ih[k];
        const float4 w4 = wh[k];
        acc += s_feat[i4.x] * w4.x;
        acc += s_feat[i4.y] * w4.y;
        acc += s_feat[i4.z] * w4.z;
        acc += s_feat[i4.w] * w4.w;
    }
    acc += hb[tid];
    hidden[(size_t)b * NH + tid] = 1.0f / (1.0f + __expf(-acc));
}

// ---------------------------------------------------------------------------
// Kernel 3: out = sigmoid(hidden @ out_w^T + out_b), 256x256x10 via WMMA.
// One wave per 16-row batch tile; N=16 (10 valid), K loop of 8 x K=32.
// ---------------------------------------------------------------------------
__global__ __launch_bounds__(32) void out_layer_wmma(
    const float* __restrict__ hidden, const float* __restrict__ ow,
    const float* __restrict__ ob, float* __restrict__ out)
{
    const int lane = threadIdx.x & 31;
    const int m    = lane & 15;
    const bool hi  = (lane >> 4) != 0;
    const int b0   = blockIdx.x * 16;
    const int mr   = (m < NO) ? m : 0;      // clamped valid row for unconditional loads

    v8f acc = {};
    #pragma unroll
    for (int k0 = 0; k0 < NH; k0 += 32) {
        v16h a, bf;
        #pragma unroll
        for (int h = 0; h < 16; ++h) {
            const int Ka = k0 + ((h < 8) ? h : h + 8) + (hi ? 8 : 0);
            a[h] = (_Float16)hidden[(size_t)(b0 + m) * NH + Ka];
            const int Kb = k0 + ((h < 8) ? h : h + 8) + (hi ? 8 : 0);
            const float raw = ow[(size_t)mr * NH + Kb];
            bf[h] = (_Float16)((m < NO) ? raw : 0.0f);
        }
        acc = __builtin_amdgcn_wmma_f32_16x16x32_f16(
            false, a, false, bf, (short)0, acc, false, false);
    }

    if (m < NO) {
        const float bo = ob[m];
        #pragma unroll
        for (int r = 0; r < 8; ++r) {
            const int bb = b0 + r + (hi ? 8 : 0);
            const float v = acc[r] + bo;
            out[(size_t)bb * NO + m] = 1.0f / (1.0f + __expf(-v));
        }
    }
}

// ---------------------------------------------------------------------------
extern "C" void kernel_launch(void* const* d_in, const int* in_sizes, int n_in,
                              void* d_out, int out_size, void* d_ws, size_t ws_size,
                              hipStream_t stream) {
    (void)in_sizes; (void)n_in; (void)out_size; (void)ws_size;
    const float* inputs   = (const float*)d_in[0];
    const float* conv_w   = (const float*)d_in[1];
    const float* conv_b   = (const float*)d_in[2];
    const int*   hidden_i = (const int*)  d_in[3];
    const float* hidden_w = (const float*)d_in[4];
    const float* hidden_b = (const float*)d_in[5];
    const float* out_w    = (const float*)d_in[6];
    const float* out_b    = (const float*)d_in[7];
    float*       out      = (float*)d_out;

    float* feats  = (float*)d_ws;                                  // 65,028,096 B
    float* hidden = (float*)((char*)d_ws + (size_t)BATCH * FEAT * 4);

    conv_pool_wmma<<<BATCH * PW, 256, 0, stream>>>(inputs, conv_w, conv_b, feats);
    gather_hidden<<<BATCH, 256, FEAT * sizeof(float), stream>>>(
        feats, hidden_i, hidden_w, hidden_b, hidden);
    out_layer_wmma<<<BATCH / 16, 32, 0, stream>>>(hidden, out_w, out_b, out);
}
